// CapsuleLayer_16449724743919
// MI455X (gfx1250) — compile-verified, matrix operand order checked by reading, not run on hardware
//
#include <hip/hip_runtime.h>

// Capsule routing layer, MI455X (gfx1250, wave32).
//
// Plan (roofline): hat is 47.2M elems -> store f16 (94MB, L2-resident).
//   K1: hat[b,i,j,d] via v_wmma_f32_16x16x32_f16 (K padded 8->32), f16 store.
//   K2 x3: one fused streaming pass over hat per routing round:
//          agreement dot (shfl over d) + logit accum + per-(b,i) softmax
//          over j (lane-local) + weighted sum into registers.
// Total traffic ~450MB -> ~20us @ 23TB/s, mostly L2 hits. Compute ~3 GFLOP.

typedef __attribute__((ext_vector_type(16))) _Float16 v16h;
typedef __attribute__((ext_vector_type(8)))  float    v8f;

#define BATCH   256
#define IN_CAPS 1152
#define IN_DIM  8
#define NUM_CAPS 10
#define DIM_VEC  16
#define JD (NUM_CAPS * DIM_VEC)   // 160

// ---------------------------------------------------------------------------
// Kernel 1: hat[b,i,j,d] = sum_e x[b,i,e] * W[i,j,e,d]   (f16 output)
// One wave per (batch-tile m, input capsule i). 10 WMMAs per block (one per j).
// A frag (16x32 f16): lanes 0-15 hold K=0..7 in packed halves a[0..7]
// (K=8..31 zero => lanes 16-31 and a[8..15] zero). B frag mirrored on N=d.
// ---------------------------------------------------------------------------
__global__ __launch_bounds__(32)
void caps_hat_wmma(const float* __restrict__ x,
                   const float* __restrict__ W,
                   _Float16*    __restrict__ hat) {
  const int m    = blockIdx.x;          // batch tile: rows m*16 .. m*16+15
  const int i    = blockIdx.y;          // input capsule
  const int lane = threadIdx.x & 31;
  const int half = lane >> 4;           // 0: K=0..7 carrier, 1: zeros
  const int l15  = lane & 15;           // M (for A) / N=d (for B)

  v16h a;
  #pragma unroll
  for (int k = 0; k < 16; ++k) a[k] = (_Float16)0.0f;
  if (half == 0) {
    const float* xp = x + ((size_t)(m * 16 + l15) * IN_CAPS + i) * IN_DIM;
    #pragma unroll
    for (int k = 0; k < IN_DIM; ++k) a[k] = (_Float16)xp[k];
  }

  const float* wp = W + (size_t)i * NUM_CAPS * IN_DIM * DIM_VEC;

  for (int j = 0; j < NUM_CAPS; ++j) {
    v16h bf;
    #pragma unroll
    for (int k = 0; k < 16; ++k) bf[k] = (_Float16)0.0f;
    if (half == 0) {
      #pragma unroll
      for (int k = 0; k < IN_DIM; ++k)
        bf[k] = (_Float16)wp[(j * IN_DIM + k) * DIM_VEC + l15];
    }
    v8f c = {};
    c = __builtin_amdgcn_wmma_f32_16x16x32_f16(
        /*neg_a=*/false, a, /*neg_b=*/false, bf,
        /*c_mod=*/(short)0, c, /*reuse_a=*/false, /*reuse_b=*/false);

    // D layout: VGPR v, lane -> (M = v + 8*half, N = l15)
    #pragma unroll
    for (int v = 0; v < 8; ++v) {
      const int b = m * 16 + v + 8 * half;
      hat[(((size_t)b * IN_CAPS + i) * NUM_CAPS + j) * DIM_VEC + l15] =
          (_Float16)c[v];
    }
  }
}

// ---------------------------------------------------------------------------
// Kernel 2: one routing round. Block = one batch element b, 256 threads.
// Lane layout: d = lane&15, each half-wave owns one i-instance (16 instances
// per block -> i strides by 16). Agreement reduces over d with shfl_xor
// (wave32: masks 1/2/4/8 stay inside the 16-lane half). Softmax over j is
// lane-local (replicated across d lanes). s[j] accumulates in registers.
//   has_v : add <hat, v_prev> agreement to logits
//   wr_L  : persist cumulative agreement (round 1)
//   rd_L  : load prior cumulative agreement (round 2)
// ---------------------------------------------------------------------------
__global__ __launch_bounds__(256)
void caps_route_round(const _Float16* __restrict__ hat,
                      const float*    __restrict__ bias,   // [IN_CAPS, NUM_CAPS]
                      float*          __restrict__ Lws,    // [B, IN_CAPS, NUM_CAPS]
                      const float*    __restrict__ vprev,  // [B, NUM_CAPS, DIM_VEC]
                      float*          __restrict__ vout,   // [B, NUM_CAPS, DIM_VEC]
                      int has_v, int wr_L, int rd_L) {
  const int b    = blockIdx.x;
  const int tid  = threadIdx.x;
  const int lane = tid & 31;
  const int wave = tid >> 5;
  const int d    = lane & 15;
  const int half = lane >> 4;
  const int inst = wave * 2 + half;      // 0..15

  __shared__ float vlds[JD];
  __shared__ float sred[16][JD];

  if (has_v && tid < JD) vlds[tid] = vprev[(size_t)b * JD + tid];
  __syncthreads();

  float s_acc[NUM_CAPS];
  #pragma unroll
  for (int j = 0; j < NUM_CAPS; ++j) s_acc[j] = 0.0f;

  for (int i = inst; i < IN_CAPS; i += 16) {
    const _Float16* hp = hat + ((size_t)b * IN_CAPS + i) * JD;
    const float*    bp = bias + (size_t)i * NUM_CAPS;
    float*          lp = Lws + ((size_t)b * IN_CAPS + i) * NUM_CAPS;

    float hv[NUM_CAPS], lg[NUM_CAPS];
    #pragma unroll
    for (int j = 0; j < NUM_CAPS; ++j) {
      const float h = (float)hp[j * DIM_VEC + d];
      hv[j] = h;
      float ag = rd_L ? lp[j] : 0.0f;
      if (has_v) {
        float p = h * vlds[j * DIM_VEC + d];
        p += __shfl_xor(p, 1, 32);
        p += __shfl_xor(p, 2, 32);
        p += __shfl_xor(p, 4, 32);
        p += __shfl_xor(p, 8, 32);   // p = <hat[b,i,j,:], vprev[b,j,:]>
        ag += p;
      }
      if (wr_L && d == 0) lp[j] = ag;   // one writer per (b,i,j)
      lg[j] = bp[j] + ag;
    }

    // softmax over j (per-lane; values replicated across the 16 d-lanes)
    float mx = lg[0];
    #pragma unroll
    for (int j = 1; j < NUM_CAPS; ++j) mx = fmaxf(mx, lg[j]);
    float den = 0.0f;
    #pragma unroll
    for (int j = 0; j < NUM_CAPS; ++j) { lg[j] = __expf(lg[j] - mx); den += lg[j]; }
    const float inv = 1.0f / den;
    #pragma unroll
    for (int j = 0; j < NUM_CAPS; ++j) s_acc[j] += (lg[j] * inv) * hv[j];
  }

  // cross-instance reduction of s[j][d]
  #pragma unroll
  for (int j = 0; j < NUM_CAPS; ++j) sred[inst][j * DIM_VEC + d] = s_acc[j];
  __syncthreads();

  if (tid < JD) {                       // tid = j*16 + d
    float s = 0.0f;
    #pragma unroll
    for (int k = 0; k < 16; ++k) s += sred[k][tid];
    // squash: scale = |s|^2 / (1+|s|^2) / |s| ; |s|^2 over d (half-wave)
    float p = s * s;
    p += __shfl_xor(p, 1, 32);
    p += __shfl_xor(p, 2, 32);
    p += __shfl_xor(p, 4, 32);
    p += __shfl_xor(p, 8, 32);
    const float scale = p / ((1.0f + p) * sqrtf(p));
    vout[(size_t)b * JD + tid] = scale * s;
  }
}

// ---------------------------------------------------------------------------
extern "C" void kernel_launch(void* const* d_in, const int* in_sizes, int n_in,
                              void* d_out, int out_size, void* d_ws, size_t ws_size,
                              hipStream_t stream) {
  const float* x    = (const float*)d_in[0];   // [B, IN_CAPS, IN_DIM]
  const float* W    = (const float*)d_in[1];   // [IN_CAPS, NUM_CAPS, IN_DIM, DIM_VEC]
  const float* bias = (const float*)d_in[2];   // [1, IN_CAPS, NUM_CAPS]
  float* out = (float*)d_out;                  // [B, NUM_CAPS, DIM_VEC]

  char* ws = (char*)d_ws;
  const size_t hat_bytes = (size_t)BATCH * IN_CAPS * NUM_CAPS * DIM_VEC * sizeof(_Float16); // 94.4 MB
  const size_t L_bytes   = (size_t)BATCH * IN_CAPS * NUM_CAPS * sizeof(float);              // 11.8 MB
  _Float16* hat = (_Float16*)ws;
  float*    Lws = (float*)(ws + hat_bytes);
  float*    v0  = (float*)(ws + hat_bytes + L_bytes);
  float*    v1  = v0 + (size_t)BATCH * JD;

  // 1) hat via WMMA (f16 store; L2-resident for the routing passes)
  caps_hat_wmma<<<dim3(16, IN_CAPS), 32, 0, stream>>>(x, W, hat);

  // 2) three fused routing rounds (one streaming pass over hat each)
  //    round 0: c = softmax(bias);             v0 = squash(s)
  //    round 1: L = <hat,v0>; c=softmax(b+L);  v1 = squash(s)   (write L)
  //    round 2: c = softmax(b + L + <hat,v1>); out = squash(s)  (read L)
  caps_route_round<<<BATCH, 256, 0, stream>>>(hat, bias, Lws, v0, v0, 0, 0, 0);
  caps_route_round<<<BATCH, 256, 0, stream>>>(hat, bias, Lws, v0, v1, 1, 1, 0);
  caps_route_round<<<BATCH, 256, 0, stream>>>(hat, bias, Lws, v1, out, 1, 0, 1);
}
